// MODEL_42726334660686
// MI455X (gfx1250) — matrix operand order repaired and checked by reading, not compiled, not run on hardware
//
#include <hip/hip_runtime.h>
#include <hip/hip_bf16.h>
#include <math.h>

// ---------------------------------------------------------------------------
// DKVMN forward for MI455X (gfx1250, wave32, WMMA).
//  (0) one-shot f32->f16 conversion of embeddings (gathered) and weights
//  (1) w_all  = softmax(q_emb @ mem_key^T)          [WMMA f32<-f16 16x16x32]
//  (2) e_all/a_all = sigmoid/tanh(qa_emb @ W^T + b) [WMMA]
//  (3) sequential memory scan per batch (LDS-resident mem, 64 blocks)
//  (4) head GEMM + logits + BCE loss                [WMMA]
// All WMMA fragments are raw f16 b128 loads: no cvt on the WMMA dep path.
// ---------------------------------------------------------------------------

typedef __attribute__((ext_vector_type(16))) _Float16 v16h;
typedef __attribute__((ext_vector_type(8)))  _Float16 v8h;
typedef __attribute__((ext_vector_type(8)))  float    v8f;

union F32Acc { v8f v; float f[8]; };

__device__ __forceinline__ v8f wmma_f16(v16h a, v16h b, v8f c) {
  // (neg_a, A, neg_b, B, c_mod, C, reuse_a, reuse_b)
  return __builtin_amdgcn_wmma_f32_16x16x32_f16(false, a, false, b, (short)0, c,
                                                false, false);
}

__device__ __forceinline__ v16h cat16(v8h a, v8h b) {
  return __builtin_shufflevector(a, b, 0, 1, 2, 3, 4, 5, 6, 7,
                                 8, 9, 10, 11, 12, 13, 14, 15);
}

// A-matrix 16x32 f16 fragment (lane row m = lane&15, group g = lane>>4).
// Halves j=0..15 hold K = k0+16g+j:
// two contiguous 8-half chunks -> two global_load_b128.
__device__ __forceinline__ v16h load_a_frag(const _Float16* rowPtr, int k0, int g) {
  v8h c0 = *reinterpret_cast<const v8h*>(rowPtr + k0 + 16 * g);
  v8h c1 = *reinterpret_cast<const v8h*>(rowPtr + k0 + 16 * g + 8);
  return cat16(c0, c1);
}

// B-matrix 32x16 f16 fragment: lane column n = lane&15, halves j hold K = k0+16g+j.
// For B = W^T with row-major W (N x K): 16 contiguous halves at W + n*K + k0 + 16g.
__device__ __forceinline__ v16h load_b_frag(const _Float16* p) {
  v8h c0 = *reinterpret_cast<const v8h*>(p);
  v8h c1 = *reinterpret_cast<const v8h*>(p + 8);
  return cat16(c0, c1);
}

__device__ __forceinline__ float softplusf(float x) {
  return log1pf(expf(-fabsf(x))) + fmaxf(x, 0.0f);
}

// ---------------------------------------------------------------------------
// Conversion kernels (run once per launch; tiny vs. 23.3 TB/s HBM).
// ---------------------------------------------------------------------------
__global__ void cvt_f16_kernel(const float* __restrict__ src,
                               _Float16* __restrict__ dst, int n) {
  const int i = blockIdx.x * blockDim.x + threadIdx.x;
  if (i < n) dst[i] = (_Float16)src[i];
}

// One block per row; blockDim == row width (128 or 256).
__global__ void gather_f16_kernel(const int* __restrict__ idx,
                                  const float* __restrict__ table,
                                  _Float16* __restrict__ dst) {
  const int row = blockIdx.x;
  const int col = threadIdx.x;
  const int w = blockDim.x;
  dst[row * w + col] = (_Float16)table[idx[row] * w + col];
}

// ---------------------------------------------------------------------------
// Kernel 1: w_all[row][64] = softmax_row( qemb_h[row] @ key_h^T )
// 256 blocks x 128 threads (4 waves); each wave: 16 rows, N=64 (4 tiles), K=128.
// ---------------------------------------------------------------------------
__global__ void dkvmn_w_kernel(const _Float16* __restrict__ qemb_h,
                               const _Float16* __restrict__ key_h,
                               float* __restrict__ w_all) {
  const int wave = threadIdx.x >> 5;
  const int lane = threadIdx.x & 31;
  const int g = lane >> 4, n = lane & 15;
  const int rowBase = blockIdx.x * 64 + wave * 16;

  const _Float16* arow = qemb_h + (rowBase + n) * 128;

  F32Acc acc[4];
#pragma unroll
  for (int nt = 0; nt < 4; ++nt)
#pragma unroll
    for (int r = 0; r < 8; ++r) acc[nt].f[r] = 0.0f;

#pragma unroll
  for (int ks = 0; ks < 4; ++ks) {
    v16h af = load_a_frag(arow, ks * 32, g);
#pragma unroll
    for (int nt = 0; nt < 4; ++nt) {
      v16h bf = load_b_frag(key_h + (nt * 16 + n) * 128 + ks * 32 + 16 * g);
      acc[nt].v = wmma_f16(af, bf, acc[nt].v);
    }
  }

  // Row-wise softmax over 64 logits: row m=r+8g lives on 16 lanes (one N each)
  // across the 4 accumulators. Reduce via shfl_xor within the 16-lane group.
#pragma unroll
  for (int r = 0; r < 8; ++r) {
    float lm = fmaxf(fmaxf(acc[0].f[r], acc[1].f[r]),
                     fmaxf(acc[2].f[r], acc[3].f[r]));
#pragma unroll
    for (int msk = 1; msk < 16; msk <<= 1) lm = fmaxf(lm, __shfl_xor(lm, msk, 32));
    float ex[4];
    float s = 0.0f;
#pragma unroll
    for (int nt = 0; nt < 4; ++nt) { ex[nt] = expf(acc[nt].f[r] - lm); s += ex[nt]; }
#pragma unroll
    for (int msk = 1; msk < 16; msk <<= 1) s += __shfl_xor(s, msk, 32);
    const float inv = 1.0f / s;
    const int row = rowBase + r + 8 * g;
#pragma unroll
    for (int nt = 0; nt < 4; ++nt) w_all[row * 64 + nt * 16 + n] = ex[nt] * inv;
  }
}

// ---------------------------------------------------------------------------
// Kernel 2: e_all = sigmoid(qa_emb @ erase_w^T + eb), a_all = tanh(qa_emb @ add_w^T + ab)
// 256 blocks x 128 threads; each wave: 16 rows, N=128 (8 tiles), K=256 (8 steps).
// ---------------------------------------------------------------------------
__global__ void dkvmn_ea_kernel(const _Float16* __restrict__ qaemb_h,
                                const _Float16* __restrict__ ew_h,
                                const float* __restrict__ ebias,
                                const _Float16* __restrict__ aw_h,
                                const float* __restrict__ abias,
                                float* __restrict__ e_all,
                                float* __restrict__ a_all) {
  const int wave = threadIdx.x >> 5;
  const int lane = threadIdx.x & 31;
  const int g = lane >> 4, n = lane & 15;
  const int rowBase = blockIdx.x * 64 + wave * 16;

  const _Float16* arow = qaemb_h + (rowBase + n) * 256;

  v16h af[8];
#pragma unroll
  for (int ks = 0; ks < 8; ++ks) af[ks] = load_a_frag(arow, ks * 32, g);

#pragma unroll
  for (int nt = 0; nt < 8; ++nt) {
    const int ng = nt * 16 + n;
    F32Acc ae, aa;
#pragma unroll
    for (int r = 0; r < 8; ++r) { ae.f[r] = 0.0f; aa.f[r] = 0.0f; }
#pragma unroll
    for (int ks = 0; ks < 8; ++ks) {
      v16h be = load_b_frag(ew_h + ng * 256 + ks * 32 + 16 * g);
      ae.v = wmma_f16(af[ks], be, ae.v);
      v16h ba = load_b_frag(aw_h + ng * 256 + ks * 32 + 16 * g);
      aa.v = wmma_f16(af[ks], ba, aa.v);
    }
    const float eb = ebias[ng], ab = abias[ng];
#pragma unroll
    for (int r = 0; r < 8; ++r) {
      const int row = rowBase + r + 8 * g;
      e_all[row * 128 + ng] = 1.0f / (1.0f + expf(-(ae.f[r] + eb)));
      a_all[row * 128 + ng] = tanhf(aa.f[r] + ab);
    }
  }
}

// ---------------------------------------------------------------------------
// Kernel 3: sequential memory scan. One block per batch (64 blocks x 128 thr).
// mem (64x128 f32 = 32 KB) lives in LDS; each thread owns one value column.
// Emits read directly as f16 for the head GEMM's A fragments.
// ---------------------------------------------------------------------------
__global__ void dkvmn_scan_kernel(const float* __restrict__ w_all,
                                  const float* __restrict__ e_all,
                                  const float* __restrict__ a_all,
                                  const float* __restrict__ mem0,
                                  _Float16* __restrict__ read_h) {
  __shared__ float memLds[64 * 128];
  __shared__ float wLds[64];
  const int b = blockIdx.x, tid = threadIdx.x;

  for (int i = tid; i < 64 * 128; i += 128) memLds[i] = mem0[i];
  __syncthreads();

  const int v = tid;  // value column 0..127 (thread-private across all slots)
  for (int t = 0; t < 256; ++t) {
    const int row = b * 256 + t;
    if (tid < 64) wLds[tid] = w_all[row * 64 + tid];
    __syncthreads();
    const float e = e_all[row * 128 + v];
    const float a = a_all[row * 128 + v];
    float rd = 0.0f;
    for (int m = 0; m < 64; ++m) {
      const float wl = wLds[m];
      const float old = memLds[m * 128 + v];
      rd += wl * old;
      memLds[m * 128 + v] = old * (1.0f - wl * e) + wl * a;
    }
    read_h[row * 128 + v] = (_Float16)rd;
    __syncthreads();
  }
}

// ---------------------------------------------------------------------------
// Kernel 4: head. feat = [read | q_emb] (K=256), h = tanh(feat @ read_w^T + rb),
// logit = h . pred_w + pb; emit sigmoid(logit), target copy, masked BCE partials.
// ---------------------------------------------------------------------------
__global__ void dkvmn_head_kernel(const _Float16* __restrict__ qemb_h,
                                  const _Float16* __restrict__ read_h,
                                  const _Float16* __restrict__ rw_h,
                                  const float* __restrict__ rb,
                                  const float* __restrict__ pw,
                                  const float* __restrict__ pb,
                                  const float* __restrict__ target,
                                  float* __restrict__ out,
                                  float* __restrict__ accum) {
  const int wave = threadIdx.x >> 5;
  const int lane = threadIdx.x & 31;
  const int g = lane >> 4, n = lane & 15;
  const int rowBase = blockIdx.x * 64 + wave * 16;

  const _Float16* rrow = read_h + (rowBase + n) * 128;
  const _Float16* qrow = qemb_h + (rowBase + n) * 128;

  v16h af[8];
#pragma unroll
  for (int ks = 0; ks < 4; ++ks) af[ks] = load_a_frag(rrow, ks * 32, g);
#pragma unroll
  for (int ks = 4; ks < 8; ++ks) af[ks] = load_a_frag(qrow, ks * 32 - 128, g);

  float p[8];
#pragma unroll
  for (int r = 0; r < 8; ++r) p[r] = 0.0f;

#pragma unroll
  for (int nt = 0; nt < 8; ++nt) {
    const int ng = nt * 16 + n;
    F32Acc acc;
#pragma unroll
    for (int r = 0; r < 8; ++r) acc.f[r] = 0.0f;
#pragma unroll
    for (int ks = 0; ks < 8; ++ks) {
      v16h bf = load_b_frag(rw_h + ng * 256 + ks * 32 + 16 * g);
      acc.v = wmma_f16(af[ks], bf, acc.v);
    }
    const float rbv = rb[ng], pwv = pw[ng];
#pragma unroll
    for (int r = 0; r < 8; ++r) p[r] += tanhf(acc.f[r] + rbv) * pwv;
  }

  // reduce h.pred_w over the 16 lanes holding a row's N slices
#pragma unroll
  for (int r = 0; r < 8; ++r)
#pragma unroll
    for (int msk = 1; msk < 16; msk <<= 1) p[r] += __shfl_xor(p[r], msk, 32);

  if (n == 0) {  // lanes 0 (rows r) and 16 (rows r+8)
    const float pbv = pb[0];
    float lsum = 0.0f, lcnt = 0.0f;
#pragma unroll
    for (int r = 0; r < 8; ++r) {
      const int row = rowBase + r + 8 * g;
      const float logit = p[r] + pbv;
      out[1 + row] = 1.0f / (1.0f + expf(-logit));
      const float tg = target[row];
      out[1 + 16384 + row] = tg;
      const float mask = (tg >= 0.0f) ? 1.0f : 0.0f;
      lsum += (tg * softplusf(-logit) + (1.0f - tg) * softplusf(logit)) * mask;
      lcnt += mask;
    }
    atomicAdd(accum, lsum);
    atomicAdd(accum + 1, lcnt);
  }
}

__global__ void dkvmn_zero_kernel(float* __restrict__ accum) {
  if (threadIdx.x < 2) accum[threadIdx.x] = 0.0f;
}

__global__ void dkvmn_final_kernel(const float* __restrict__ accum,
                                   float* __restrict__ out) {
  out[0] = accum[0] / fmaxf(accum[1], 1.0f);
}

// ---------------------------------------------------------------------------
extern "C" void kernel_launch(void* const* d_in, const int* in_sizes, int n_in,
                              void* d_out, int out_size, void* d_ws, size_t ws_size,
                              hipStream_t stream) {
  (void)in_sizes; (void)n_in; (void)out_size; (void)ws_size;
  const int*   q_data    = (const int*)d_in[0];
  const int*   qa_data   = (const int*)d_in[1];
  const float* target    = (const float*)d_in[2];
  const float* q_embed_w = (const float*)d_in[3];
  const float* qa_embed_w= (const float*)d_in[4];
  const float* mem_key   = (const float*)d_in[5];
  const float* mem_v0    = (const float*)d_in[6];
  const float* erase_w   = (const float*)d_in[7];
  const float* erase_b   = (const float*)d_in[8];
  const float* add_w     = (const float*)d_in[9];
  const float* add_b     = (const float*)d_in[10];
  const float* read_w    = (const float*)d_in[11];
  const float* read_b    = (const float*)d_in[12];
  const float* pred_w    = (const float*)d_in[13];
  const float* pred_b    = (const float*)d_in[14];
  float* out = (float*)d_out;

  // Workspace layout (all chunks 16B-aligned by construction).
  char* ws = (char*)d_ws;
  float* w_all = (float*)ws;                 ws += (size_t)16384 * 64 * 4;   // 4 MB
  float* e_all = (float*)ws;                 ws += (size_t)16384 * 128 * 4;  // 8 MB
  float* a_all = (float*)ws;                 ws += (size_t)16384 * 128 * 4;  // 8 MB
  _Float16* qemb_h  = (_Float16*)ws;         ws += (size_t)16384 * 128 * 2;  // 4 MB
  _Float16* qaemb_h = (_Float16*)ws;         ws += (size_t)16384 * 256 * 2;  // 8 MB
  _Float16* read_h  = (_Float16*)ws;         ws += (size_t)16384 * 128 * 2;  // 4 MB
  _Float16* key_h   = (_Float16*)ws;         ws += (size_t)64 * 128 * 2;
  _Float16* ew_h    = (_Float16*)ws;         ws += (size_t)128 * 256 * 2;
  _Float16* aw_h    = (_Float16*)ws;         ws += (size_t)128 * 256 * 2;
  _Float16* rw_h    = (_Float16*)ws;         ws += (size_t)128 * 256 * 2;
  float* accum = (float*)ws;

  // (0) one-shot conversions / gathers to f16
  gather_f16_kernel<<<16384, 128, 0, stream>>>(q_data, q_embed_w, qemb_h);
  gather_f16_kernel<<<16384, 256, 0, stream>>>(qa_data, qa_embed_w, qaemb_h);
  cvt_f16_kernel<<<(64 * 128 + 255) / 256, 256, 0, stream>>>(mem_key, key_h, 64 * 128);
  cvt_f16_kernel<<<(128 * 256 + 255) / 256, 256, 0, stream>>>(erase_w, ew_h, 128 * 256);
  cvt_f16_kernel<<<(128 * 256 + 255) / 256, 256, 0, stream>>>(add_w, aw_h, 128 * 256);
  cvt_f16_kernel<<<(128 * 256 + 255) / 256, 256, 0, stream>>>(read_w, rw_h, 128 * 256);
  dkvmn_zero_kernel<<<1, 32, 0, stream>>>(accum);

  // (1)..(4)
  dkvmn_w_kernel<<<256, 128, 0, stream>>>(qemb_h, key_h, w_all);
  dkvmn_ea_kernel<<<256, 128, 0, stream>>>(qaemb_h, ew_h, erase_b, aw_h, add_b,
                                           e_all, a_all);
  dkvmn_scan_kernel<<<64, 128, 0, stream>>>(w_all, e_all, a_all, mem_v0, read_h);
  dkvmn_head_kernel<<<256, 128, 0, stream>>>(qemb_h, read_h, rw_h, read_b,
                                             pred_w, pred_b, target, out, accum);
  dkvmn_final_kernel<<<1, 1, 0, stream>>>(accum, out);
}
